// BCMPLayer1_88467736363033
// MI455X (gfx1250) — compile-verified, matrix operand order checked by reading, not run on hardware
//
#include <hip/hip_runtime.h>

// ---------------------------------------------------------------------------
// BCMP layer for MI455X (gfx1250, wave32, WMMA).
//
// out = ( ahat(x@WX) + bhat(bc@WZ) + ahat(bhat(bc@Walpha)) ) / 3
//     = ahat( x@WX + bhat(bc@Walpha) ) / 3  +  bhat(bc@WZ) / 3      (linearity)
//
// Roofline: big GEMM is ~3.3 GFLOP but only ~100 MB traffic -> memory bound;
// edge propagation (1.6 GB irregular) dominates, and H/Agg (51 MB each) fit
// in the 192 MB L2, so gathers/atomics run at L2 bandwidth.
// GEMM uses V_WMMA_F32_16X16X32_BF16 with a bf16 hi/lo split (3 WMMAs per
// tile-step) for near-f32 accuracy; compute is free at this arithmetic
// intensity.
// ---------------------------------------------------------------------------

#define DIM 128

typedef __attribute__((ext_vector_type(16))) __bf16 v16bf;
typedef __attribute__((ext_vector_type(8)))  float  v8f;

// ---------------- degree / count helpers ----------------------------------

__global__ __launch_bounds__(256) void k_count(const int* __restrict__ idx,
                                               float* __restrict__ cnt, int n) {
    int i = blockIdx.x * blockDim.x + threadIdx.x;
    if (i < n) atomicAdd(&cnt[idx[i]], 1.0f);
}

__global__ __launch_bounds__(256) void k_rsqrt1p(float* __restrict__ v, int n) {
    int i = blockIdx.x * blockDim.x + threadIdx.x;
    if (i < n) v[i] = rsqrtf(1.0f + v[i]);
}

// ---------------- big GEMM: H = X @ W via WMMA bf16-split ------------------
// One workgroup = 8 waves; wave w computes rows [tile*16, tile*16+16) for all
// 128 output columns (8 N-tiles of 16). K = 128 -> 4 steps of K=32.
// W (row-major [K][N]) is split into bf16 hi/lo and TRANSPOSED into LDS as
// [N][K] so B-fragment loads are contiguous.
//
// Documented 16-bit fragment layouts (CDNA5 ISA):
//   A 16x32: lane<16 -> row=lane,  elems 0..7 = K0..7,  8..15 = K16..23
//            lane>=16 -> row=lane-16, elems 0..7 = K8..15, 8..15 = K24..31
//   B 32x16: mirrored (lane selects column N, same K split)
//   C 16x16 f32: elem r of lane l -> M = r + 8*(l>=16), N = l%16
__global__ __launch_bounds__(256) void k_gemm_wmma(
    const float* __restrict__ X, const float* __restrict__ W,
    float* __restrict__ H, int nrows) {
    __shared__ __bf16 Wt_hi[DIM * DIM];
    __shared__ __bf16 Wt_lo[DIM * DIM];

    for (int idx = threadIdx.x; idx < DIM * DIM; idx += 256) {
        int k = idx >> 7, n = idx & 127;        // W row-major [k][n]
        float w = W[idx];
        __bf16 whi = (__bf16)w;
        __bf16 wlo = (__bf16)(w - (float)whi);
        Wt_hi[n * DIM + k] = whi;               // transposed: [n][k]
        Wt_lo[n * DIM + k] = wlo;
    }
    __syncthreads();

    int wave = threadIdx.x >> 5;
    int lane = threadIdx.x & 31;
    int tile = blockIdx.x * 8 + wave;           // 16-row tile index
    if (tile * 16 >= nrows) return;

    int r0      = tile * 16;
    int lrow    = lane & 15;                    // M (for A) / N (for B,C)
    bool hiHalf = lane >= 16;
    int kA0 = hiHalf ? 8 : 0;                   // K offset of elems 0..7
    int kA1 = hiHalf ? 24 : 16;                 // K offset of elems 8..15

    const float* xrow = X + (size_t)(r0 + lrow) * DIM;

    v8f acc[8];
#pragma unroll
    for (int nt = 0; nt < 8; ++nt) acc[nt] = (v8f)(0.0f);

#pragma unroll
    for (int kb = 0; kb < 4; ++kb) {
        int kbase = kb * 32;
        v16bf a_hi, a_lo;
#pragma unroll
        for (int j = 0; j < 8; ++j) {
            float x0 = xrow[kbase + kA0 + j];
            float x1 = xrow[kbase + kA1 + j];
            __bf16 h0 = (__bf16)x0;
            __bf16 h1 = (__bf16)x1;
            a_hi[j]     = h0;
            a_hi[8 + j] = h1;
            a_lo[j]     = (__bf16)(x0 - (float)h0);
            a_lo[8 + j] = (__bf16)(x1 - (float)h1);
        }
#pragma unroll
        for (int nt = 0; nt < 8; ++nt) {
            int col = nt * 16 + lrow;
            const __bf16* bh = &Wt_hi[col * DIM + kbase];
            const __bf16* bl = &Wt_lo[col * DIM + kbase];
            v16bf b_hi, b_lo;
#pragma unroll
            for (int j = 0; j < 8; ++j) {
                b_hi[j]     = bh[kA0 + j];
                b_hi[8 + j] = bh[kA1 + j];
                b_lo[j]     = bl[kA0 + j];
                b_lo[8 + j] = bl[kA1 + j];
            }
            // hi*hi + hi*lo + lo*hi  (drop lo*lo): ~2^-16 relative error
            acc[nt] = __builtin_amdgcn_wmma_f32_16x16x32_bf16(
                false, a_hi, false, b_hi, (short)0, acc[nt], false, false);
            acc[nt] = __builtin_amdgcn_wmma_f32_16x16x32_bf16(
                false, a_hi, false, b_lo, (short)0, acc[nt], false, false);
            acc[nt] = __builtin_amdgcn_wmma_f32_16x16x32_bf16(
                false, a_lo, false, b_hi, (short)0, acc[nt], false, false);
        }
    }

    int mbase = hiHalf ? 8 : 0;
#pragma unroll
    for (int nt = 0; nt < 8; ++nt) {
        float* hout = H + (size_t)r0 * DIM + nt * 16 + lrow;
#pragma unroll
        for (int r = 0; r < 8; ++r)
            hout[(size_t)(mbase + r) * DIM] = acc[nt][r];
    }
}

// ---------------- small GEMM (M=1000): plain f32, exact -------------------

__global__ __launch_bounds__(256) void k_small_gemm(
    const float* __restrict__ A, const float* __restrict__ W,
    float* __restrict__ Out, int rows) {
    int gid = blockIdx.x * blockDim.x + threadIdx.x;
    if (gid >= rows * DIM) return;
    int r = gid >> 7, c = gid & 127;
    const float* a = A + (size_t)r * DIM;
    float s = 0.0f;
#pragma unroll 16
    for (int k = 0; k < DIM; ++k) s = fmaf(a[k], W[k * DIM + c], s);
    Out[gid] = s;
}

// ---------------- H += bhat(G3) -------------------------------------------

__global__ __launch_bounds__(256) void k_bhat_add(
    float* __restrict__ H, const float* __restrict__ G,
    const float* __restrict__ dcol, const int* __restrict__ assign,
    int n, int m) {
    size_t gid = (size_t)blockIdx.x * blockDim.x + threadIdx.x;
    if (gid >= (size_t)n * DIM) return;
    int i = (int)(gid >> 7), c = (int)(gid & 127);
    int a = assign[i];
    float v = dcol[a] * G[(size_t)a * DIM + c];
    float drow = 1.0f;
    if (i < m) { v += dcol[i] * G[(size_t)i * DIM + c]; drow = 0.70710678118654752f; }
    H[gid] += drow * v;
}

// ---------------- edge propagation: Agg[row] += d[col]*H[col] -------------
// One wave per edge; each lane moves a float4 of channels (index loads
// amortized 4x per lane). H/Agg live in L2 (51 MB each << 192 MB).

__global__ __launch_bounds__(256) void k_edge(
    const int* __restrict__ row, const int* __restrict__ col,
    const float* __restrict__ d, const float* __restrict__ H,
    float* __restrict__ Agg, long long nedges) {
    size_t gid = (size_t)blockIdx.x * blockDim.x + threadIdx.x;
    size_t e = gid >> 5;
    if (e >= (size_t)nedges) return;
    int c4 = (int)(gid & 31) * 4;
    int ce = col[e];
    int re = row[e];
    float dc = d[ce];
    const float4 h = *(const float4*)(H + (size_t)ce * DIM + c4);
    float* ag = Agg + (size_t)re * DIM + c4;
    atomicAdd(ag + 0, dc * h.x);
    atomicAdd(ag + 1, dc * h.y);
    atomicAdd(ag + 2, dc * h.z);
    atomicAdd(ag + 3, dc * h.w);
}

// ---------------- finalize: out = (d*(Agg + d*H) + bhat(G2)) / 3 ----------

__global__ __launch_bounds__(256) void k_finalize(
    const float* __restrict__ H, const float* __restrict__ Agg,
    const float* __restrict__ d, const float* __restrict__ G2,
    const float* __restrict__ dcol, const int* __restrict__ assign,
    float* __restrict__ out, int n, int m) {
    size_t gid = (size_t)blockIdx.x * blockDim.x + threadIdx.x;
    if (gid >= (size_t)n * DIM) return;
    int i = (int)(gid >> 7), c = (int)(gid & 127);
    float di = d[i];
    float part1 = di * (Agg[gid] + di * H[gid]);
    int a = assign[i];
    float v = dcol[a] * G2[(size_t)a * DIM + c];
    float drow = 1.0f;
    if (i < m) { v += dcol[i] * G2[(size_t)i * DIM + c]; drow = 0.70710678118654752f; }
    out[gid] = (part1 + drow * v) * (1.0f / 3.0f);
}

// ---------------------------------------------------------------------------

extern "C" void kernel_launch(void* const* d_in, const int* in_sizes, int n_in,
                              void* d_out, int out_size, void* d_ws, size_t ws_size,
                              hipStream_t stream) {
    const float* x      = (const float*)d_in[0];
    const int*   eidx   = (const int*)d_in[1];
    const float* bc     = (const float*)d_in[2];
    const int*   assign = (const int*)d_in[3];
    const float* WX     = (const float*)d_in[4];
    const float* WZ     = (const float*)d_in[5];
    const float* Walpha = (const float*)d_in[6];
    float* out = (float*)d_out;

    const int N = in_sizes[0] / DIM;
    const int E = in_sizes[1] / 2;
    const int M = in_sizes[2] / DIM;

    const int* row = eidx;        // edge_index[0]
    const int* col = eidx + E;    // edge_index[1]

    // workspace layout (floats)
    float* ws   = (float*)d_ws;
    float* H    = ws;                               // N*128
    float* Agg  = H   + (size_t)N * DIM;            // N*128
    float* G2   = Agg + (size_t)N * DIM;            // M*128
    float* G3   = G2  + (size_t)M * DIM;            // M*128
    float* dbuf = G3  + (size_t)M * DIM;            // N
    float* dcol = dbuf + (size_t)N;                 // M

    // 1. zero accumulators (capture-safe async memsets)
    hipMemsetAsync(dbuf, 0, (size_t)N * sizeof(float), stream);
    hipMemsetAsync(dcol, 0, (size_t)M * sizeof(float), stream);
    hipMemsetAsync(Agg, 0, (size_t)N * DIM * sizeof(float), stream);

    // 2. degree / cluster counts
    k_count<<<(E + 255) / 256, 256, 0, stream>>>(col, dbuf, E);
    k_count<<<(N + 255) / 256, 256, 0, stream>>>(assign, dcol, N);

    // 3. d = (1+deg)^-1/2, dcol = (1+cnt)^-1/2
    k_rsqrt1p<<<(N + 255) / 256, 256, 0, stream>>>(dbuf, N);
    k_rsqrt1p<<<(M + 255) / 256, 256, 0, stream>>>(dcol, M);

    // 4. H = x @ WX  (WMMA bf16-split; 8 waves * 16 rows = 128 rows / block)
    {
        int tiles  = (N + 15) / 16;
        int blocks = (tiles + 7) / 8;
        k_gemm_wmma<<<blocks, 256, 0, stream>>>(x, WX, H, N);
    }

    // 5. small GEMMs (exact f32)
    k_small_gemm<<<(M * DIM + 255) / 256, 256, 0, stream>>>(bc, Walpha, G3, M);
    k_small_gemm<<<(M * DIM + 255) / 256, 256, 0, stream>>>(bc, WZ, G2, M);

    // 6. H += bhat(G3)   (fuse the two ahat passes into one via linearity)
    {
        size_t t = (size_t)N * DIM;
        k_bhat_add<<<(unsigned)((t + 255) / 256), 256, 0, stream>>>(H, G3, dcol, assign, N, M);
    }

    // 7. Agg[row] += d[col] * H[col]   (single fused propagation)
    {
        size_t t = (size_t)E * 32;
        k_edge<<<(unsigned)((t + 255) / 256), 256, 0, stream>>>(row, col, dbuf, H, Agg, E);
    }

    // 8. out = (d*(Agg + d*H) + bhat(G2)) / 3
    {
        size_t t = (size_t)N * DIM;
        k_finalize<<<(unsigned)((t + 255) / 256), 256, 0, stream>>>(H, Agg, dbuf, G2, dcol, assign, out, N, M);
    }
}
